// MiniGPT_14224931684399
// MI455X (gfx1250) — matrix operand order repaired and checked by reading, not compiled, not run on hardware
//
#include <hip/hip_runtime.h>
#include <hip/hip_bf16.h>
#include <math.h>

typedef __attribute__((ext_vector_type(16))) _Float16 v16h;
typedef __attribute__((ext_vector_type(8)))  float    v8f;

#define B_  2
#define T_  1024
#define E_  1024
#define H_  16
#define HD_ 64
#define L_  6
#define V_  50257
#define M_  (B_*T_)   // 2048 rows of activations

// ---------------------------------------------------------------------------
// WMMA helpers (CDNA5 gfx1250, wave32, 16x16x32 f16 -> f32)
// ---------------------------------------------------------------------------
__device__ __forceinline__ v8f wmma16x16x32(v16h a, v16h b, v8f c) {
  // (neg_a, A, neg_b, B, c_mod, C, reuse_a, reuse_b)
  return __builtin_amdgcn_wmma_f32_16x16x32_f16(false, a, false, b, (short)0, c,
                                                false, false);
}

// A fragment: 16x32 f16 from row-major LDS tile (ld in halves).
// Per lane: row = lane&15, halves K = hi..hi+7 and 16+hi..16+hi+7 (hi = 8*(lane>=16))
// -> two contiguous 16-byte chunks -> 2x ds_load_b128.
__device__ __forceinline__ v16h frag_a(const _Float16* s, int ld, int lane) {
  const int row = lane & 15;
  const int hi  = (lane >> 4) * 8;
  v16h a;
  *((float4*)&a)       = *(const float4*)(s + row * ld + hi);
  *(((float4*)&a) + 1) = *(const float4*)(s + row * ld + 16 + hi);
  return a;
}

// B fragment from LDS stored transposed as [n][k] (ld in halves).
// Per lane: col = lane&15, halves K = hi..hi+7 and 16+hi..16+hi+7 (hi = 8*(lane>=16))
// -> 2x ds_load_b128.
__device__ __forceinline__ v16h frag_bT(const _Float16* s, int ld, int lane) {
  const int col = lane & 15;
  const int hi  = (lane >> 4) * 8;
  v16h b;
  *((float4*)&b)       = *(const float4*)(s + col * ld + hi);
  *(((float4*)&b) + 1) = *(const float4*)(s + col * ld + 16 + hi);
  return b;
}

// ---------------------------------------------------------------------------
// WMMA GEMM: C[M,N] = epi(A[M,K] * W[K,N])
// Block tile 128x128, BK=32, double-buffered LDS, 8 waves (4 row x 2 col),
// each wave owns a 32x64 output (8 accumulators, 8 WMMA per k-step).
// Weight tile staged TRANSPOSED so B fragments are contiguous b128 loads.
// Grid: blockIdx.x = M tile (fastest) so concurrent blocks share the same
// weight stripe through L2 -- weights (incl. 206MB Wlm) stream from HBM once.
// ---------------------------------------------------------------------------
enum { EPI_NONE = 0, EPI_BIAS = 1, EPI_BIAS_RES = 2, EPI_BIAS_GELU = 3 };

template <int EPI>
__global__ __launch_bounds__(256) void gemm_wmma(
    const float* __restrict__ A, const float* __restrict__ W,
    const float* __restrict__ bias, const float* res,
    float* C, int M, int N, int K)
{
  __shared__ _Float16 As[2][128 * 40];   // 128x32 act tile, padded stride 40
  __shared__ _Float16 Bt[2][128 * 40];   // 128(n) x 32(k) transposed weight tile
  const int tid  = threadIdx.x;
  const int lane = tid & 31;
  const int wv   = tid >> 5;
  const int wm   = wv & 3;               // wave row group (32 rows each)
  const int wn   = wv >> 2;              // wave col group (64 cols each)
  const int bm   = blockIdx.x * 128;     // M tile (fast dim -> L2 weight reuse)
  const int bn   = blockIdx.y * 128;     // N tile
  v8f acc[2][4] = {};

  const int ar  = tid >> 1;              // A stage: row 0..127
  const int acb = (tid & 1) * 16;        //          col base 0/16
  const int wr  = tid >> 3;              // W stage: k-row 0..31
  const int wcb = (tid & 7) * 16;        //          n base 0..112

  auto stageA = [&](int kt, int buf) {
    const float* ap = A + (size_t)(bm + ar) * K + kt + acb;
    _Float16* d = &As[buf][ar * 40 + acb];
#pragma unroll
    for (int j = 0; j < 4; ++j) {
      float4 f = *(const float4*)(ap + j * 4);
      d[j*4+0]=(_Float16)f.x; d[j*4+1]=(_Float16)f.y;
      d[j*4+2]=(_Float16)f.z; d[j*4+3]=(_Float16)f.w;
    }
  };
  auto stageB = [&](int kt, int buf) {   // transpose while staging
    const float* wp = W + (size_t)(kt + wr) * N;
#pragma unroll
    for (int j = 0; j < 16; ++j) {
      int gn = bn + wcb + j;
      _Float16 hv = (gn < N) ? (_Float16)wp[gn] : (_Float16)0.0f;
      Bt[buf][(wcb + j) * 40 + wr] = hv;
    }
  };

  stageA(0, 0);
  stageB(0, 0);
  __syncthreads();

  int cur = 0;
  for (int kt = 0; kt < K; kt += 32) {
    const int nxt = cur ^ 1;
    if (kt + 32 < K) {
      stageA(kt + 32, nxt);
      stageB(kt + 32, nxt);
      if (kt + 64 < K) {  // global_prefetch_b8 two tiles ahead
        __builtin_prefetch(A + (size_t)(bm + ar) * K + kt + 64 + acb, 0, 1);
        __builtin_prefetch(W + (size_t)(kt + 64 + wr) * N + bn + wcb, 0, 1);
      }
    }
    v16h a0 = frag_a(&As[cur][(wm * 32     ) * 40], 40, lane);
    v16h a1 = frag_a(&As[cur][(wm * 32 + 16) * 40], 40, lane);
    v16h b0 = frag_bT(&Bt[cur][(wn * 64     ) * 40], 40, lane);
    v16h b1 = frag_bT(&Bt[cur][(wn * 64 + 16) * 40], 40, lane);
    v16h b2 = frag_bT(&Bt[cur][(wn * 64 + 32) * 40], 40, lane);
    v16h b3 = frag_bT(&Bt[cur][(wn * 64 + 48) * 40], 40, lane);
    acc[0][0] = wmma16x16x32(a0, b0, acc[0][0]);
    acc[0][1] = wmma16x16x32(a0, b1, acc[0][1]);
    acc[0][2] = wmma16x16x32(a0, b2, acc[0][2]);
    acc[0][3] = wmma16x16x32(a0, b3, acc[0][3]);
    acc[1][0] = wmma16x16x32(a1, b0, acc[1][0]);
    acc[1][1] = wmma16x16x32(a1, b1, acc[1][1]);
    acc[1][2] = wmma16x16x32(a1, b2, acc[1][2]);
    acc[1][3] = wmma16x16x32(a1, b3, acc[1][3]);
    __syncthreads();
    cur = nxt;
  }

  // epilogue: C layout -> row = r + 8*(lane>=16), col = lane&15
  const int colL = lane & 15;
  const int hi8  = (lane >> 4) << 3;
#pragma unroll
  for (int mt = 0; mt < 2; ++mt) {
#pragma unroll
    for (int r = 0; r < 8; ++r) {
      int grow = bm + wm * 32 + mt * 16 + hi8 + r;
#pragma unroll
      for (int t = 0; t < 4; ++t) {
        int gcol = bn + wn * 64 + t * 16 + colL;
        if (gcol < N) {
          float val = acc[mt][t][r];
          if (EPI != EPI_NONE)      val += bias[gcol];
          if (EPI == EPI_BIAS_GELU) val = 0.5f * val * (1.0f + erff(val * 0.70710678118654752f));
          if (EPI == EPI_BIAS_RES)  val += res[(size_t)grow * N + gcol];
          C[(size_t)grow * N + gcol] = val;
        }
      }
    }
  }
}

// ---------------------------------------------------------------------------
// Flash attention: one workgroup = one (b,h) x 64-query tile.
// S = Q K^T via WMMA, online softmax (4 threads/row), O += P V via WMMA.
// q/k/v/o stored [B,T,E]; head h occupies columns h*64..h*64+63.
// K tile kept [key][dim] (already the [n][k] layout for B = K^T);
// V tile staged transposed [dim][key] so P*V B-fragments are contiguous.
// ---------------------------------------------------------------------------
__global__ __launch_bounds__(256) void attn_wmma(
    const float* __restrict__ q, const float* __restrict__ k,
    const float* __restrict__ v, float* __restrict__ o)
{
  const int qt  = blockIdx.x;        // query tile 0..15
  const int bh  = blockIdx.y;        // 0..31
  const int b   = bh >> 4;
  const int h   = bh & 15;
  const int tid = threadIdx.x, lane = tid & 31, wv = tid >> 5;
  const int wm  = wv & 3, wn = wv >> 2;

  __shared__ _Float16 Qs[64 * 72];   // [query][dim]
  __shared__ _Float16 Ks[64 * 72];   // [key][dim]   == [n][k] for B = K^T
  __shared__ _Float16 Vt[64 * 72];   // [dim][key]   == [n][k] for B = V
  __shared__ _Float16 Ps[64 * 72];   // [query][key]
  __shared__ float    Ss[64 * 65];
  __shared__ float    mrow[64], lrow[64], arow[64];
  __shared__ float    redM[64 * 4], redS[64 * 4];

  const size_t base = ((size_t)b * T_) * E_ + (size_t)h * HD_;
  const int lr_ = tid >> 2;          // staging row 0..63
  const int lc_ = (tid & 3) * 16;    // staging col base
  const int srow = tid >> 2;         // softmax row 0..63
  const int sseg = tid & 3;          // softmax 16-col segment

  {  // load Q tile once (64x64)
    const float* p = q + base + (size_t)(qt * 64 + lr_) * E_ + lc_;
    _Float16* d = &Qs[lr_ * 72 + lc_];
#pragma unroll
    for (int j = 0; j < 4; ++j) {
      float4 f = *(const float4*)(p + j * 4);
      d[j*4+0]=(_Float16)f.x; d[j*4+1]=(_Float16)f.y;
      d[j*4+2]=(_Float16)f.z; d[j*4+3]=(_Float16)f.w;
    }
  }
  if (tid < 64) { mrow[tid] = -3.0e38f; lrow[tid] = 0.0f; }
  v8f c0 = {}; v8f c1 = {};
  __syncthreads();

  for (int kt = 0; kt <= qt; ++kt) {
    {  // stage K (row-major) and V (transposed) tiles
      const float* pk = k + base + (size_t)(kt * 64 + lr_) * E_ + lc_;
      const float* pv = v + base + (size_t)(kt * 64 + lr_) * E_ + lc_;
      _Float16* dk = &Ks[lr_ * 72 + lc_];
#pragma unroll
      for (int j = 0; j < 4; ++j) {
        float4 fk = *(const float4*)(pk + j * 4);
        float4 fv = *(const float4*)(pv + j * 4);
        dk[j*4+0]=(_Float16)fk.x; dk[j*4+1]=(_Float16)fk.y;
        dk[j*4+2]=(_Float16)fk.z; dk[j*4+3]=(_Float16)fk.w;
        Vt[(lc_ + j*4 + 0) * 72 + lr_] = (_Float16)fv.x;
        Vt[(lc_ + j*4 + 1) * 72 + lr_] = (_Float16)fv.y;
        Vt[(lc_ + j*4 + 2) * 72 + lr_] = (_Float16)fv.z;
        Vt[(lc_ + j*4 + 3) * 72 + lr_] = (_Float16)fv.w;
      }
    }
    __syncthreads();

    // S = Q * K^T  (64x64): 2 K-steps of 32 over head dim
    v8f s0 = {}; v8f s1 = {};
#pragma unroll
    for (int ks = 0; ks < 64; ks += 32) {
      v16h aq  = frag_a (&Qs[wm * 16 * 72 + ks], 72, lane);
      v16h bk0 = frag_bT(&Ks[(wn * 32     ) * 72 + ks], 72, lane);
      v16h bk1 = frag_bT(&Ks[(wn * 32 + 16) * 72 + ks], 72, lane);
      s0 = wmma16x16x32(aq, bk0, s0);
      s1 = wmma16x16x32(aq, bk1, s1);
    }
    {  // scale (HD^-0.5 = 0.125), causal mask, spill to LDS
      const int colL = lane & 15;
      const int hi8  = (lane >> 4) * 8;
#pragma unroll
      for (int r = 0; r < 8; ++r) {
        int lrr = wm * 16 + hi8 + r;
        int gq  = qt * 64 + lrr;
#pragma unroll
        for (int t = 0; t < 2; ++t) {
          int lcc = wn * 32 + t * 16 + colL;
          int gk  = kt * 64 + lcc;
          float sv = ((t == 0) ? s0[r] : s1[r]) * 0.125f;
          if (gk > gq) sv = -3.0e38f;
          Ss[lrr * 65 + lcc] = sv;
        }
      }
    }
    __syncthreads();

    // online softmax: 4 threads per row
    {
      float pm = -3.0e38f;
#pragma unroll
      for (int c2 = 0; c2 < 16; ++c2)
        pm = fmaxf(pm, Ss[srow * 65 + sseg * 16 + c2]);
      redM[srow * 4 + sseg] = pm;
    }
    __syncthreads();
    if (tid < 64) {
      float mo = mrow[tid];
      float mx = fmaxf(fmaxf(redM[tid*4+0], redM[tid*4+1]),
                       fmaxf(redM[tid*4+2], fmaxf(redM[tid*4+3], mo)));
      arow[tid] = __expf(mo - mx);
      mrow[tid] = mx;
    }
    __syncthreads();
    {
      float mx = mrow[srow];
      float sum = 0.0f;
#pragma unroll
      for (int c2 = 0; c2 < 16; ++c2) {
        int cc = sseg * 16 + c2;
        float p = __expf(Ss[srow * 65 + cc] - mx);
        Ps[srow * 72 + cc] = (_Float16)p;
        sum += p;
      }
      redS[srow * 4 + sseg] = sum;
    }
    __syncthreads();
    if (tid < 64) {
      float sum = redS[tid*4+0] + redS[tid*4+1] + redS[tid*4+2] + redS[tid*4+3];
      lrow[tid] = lrow[tid] * arow[tid] + sum;
    }
    __syncthreads();

    {  // rescale running O accumulators by alpha(row)
      const int hi8 = (lane >> 4) * 8;
#pragma unroll
      for (int r = 0; r < 8; ++r) {
        float al = arow[wm * 16 + hi8 + r];
        c0[r] *= al; c1[r] *= al;
      }
    }
#pragma unroll
    for (int ks = 0; ks < 64; ks += 32) {  // O += P * V
      v16h ap  = frag_a (&Ps[wm * 16 * 72 + ks], 72, lane);
      v16h bv0 = frag_bT(&Vt[(wn * 32     ) * 72 + ks], 72, lane);
      v16h bv1 = frag_bT(&Vt[(wn * 32 + 16) * 72 + ks], 72, lane);
      c0 = wmma16x16x32(ap, bv0, c0);
      c1 = wmma16x16x32(ap, bv1, c1);
    }
    __syncthreads();
  }

  {  // normalize by row sums and store
    const int colL = lane & 15;
    const int hi8  = (lane >> 4) * 8;
#pragma unroll
    for (int r = 0; r < 8; ++r) {
      int lrr = wm * 16 + hi8 + r;
      float inv = 1.0f / lrow[lrr];
      size_t rowoff = base + (size_t)(qt * 64 + lrr) * E_;
      o[rowoff + wn * 32 + colL]      = c0[r] * inv;
      o[rowoff + wn * 32 + 16 + colL] = c1[r] * inv;
    }
  }
}

// ---------------------------------------------------------------------------
// LayerNorm: one block per row of E_ elements (biased variance, eps 1e-5)
// ---------------------------------------------------------------------------
__global__ __launch_bounds__(256) void layernorm_k(
    const float* __restrict__ x, const float* __restrict__ g,
    const float* __restrict__ bta, float* __restrict__ y)
{
  const int row = blockIdx.x;
  const int tid = threadIdx.x;
  const float* xr = x + (size_t)row * E_;
  float* yr = y + (size_t)row * E_;
  __shared__ float rs[256], rq[256];
  float s = 0.f, s2 = 0.f;
  for (int i = tid; i < E_; i += 256) { float vv = xr[i]; s += vv; s2 += vv * vv; }
  rs[tid] = s; rq[tid] = s2;
  __syncthreads();
  for (int st = 128; st > 0; st >>= 1) {
    if (tid < st) { rs[tid] += rs[tid + st]; rq[tid] += rq[tid + st]; }
    __syncthreads();
  }
  float mean = rs[0] * (1.0f / E_);
  float var  = rq[0] * (1.0f / E_) - mean * mean;
  float rstd = rsqrtf(var + 1e-5f);
  for (int i = tid; i < E_; i += 256)
    yr[i] = (xr[i] - mean) * rstd * g[i] + bta[i];
}

// ---------------------------------------------------------------------------
// Embedding: x[b,t,:] = tok_emb[idx[b,t],:] + pos_emb[t,:]
// ---------------------------------------------------------------------------
__global__ __launch_bounds__(256) void embed_k(
    const int* __restrict__ idx, const float* __restrict__ tok,
    const float* __restrict__ pos, float* __restrict__ x)
{
  const int row = blockIdx.x;         // 0..B*T-1
  const int t   = row % T_;
  const int tid = threadIdx.x;
  const size_t tokoff = (size_t)idx[row] * E_;
  for (int i = tid; i < E_; i += 256)
    x[(size_t)row * E_ + i] = tok[tokoff + i] + pos[(size_t)t * E_ + i];
}

// ---------------------------------------------------------------------------
extern "C" void kernel_launch(void* const* d_in, const int* in_sizes, int n_in,
                              void* d_out, int out_size, void* d_ws, size_t ws_size,
                              hipStream_t stream) {
  (void)in_sizes; (void)n_in; (void)out_size; (void)ws_size;
  const int*   idx = (const int*)  d_in[0];
  const float* tok = (const float*)d_in[1];
  const float* pos = (const float*)d_in[2];
  const float* Wq  = (const float*)d_in[3];
  const float* Wk  = (const float*)d_in[4];
  const float* Wv  = (const float*)d_in[5];
  const float* Wo  = (const float*)d_in[6];
  const float* bo  = (const float*)d_in[7];
  const float* g1  = (const float*)d_in[8];
  const float* be1 = (const float*)d_in[9];
  const float* g2  = (const float*)d_in[10];
  const float* be2 = (const float*)d_in[11];
  const float* W1  = (const float*)d_in[12];
  const float* b1  = (const float*)d_in[13];
  const float* W2  = (const float*)d_in[14];
  const float* b2  = (const float*)d_in[15];
  const float* gf  = (const float*)d_in[16];
  const float* bf  = (const float*)d_in[17];
  const float* Wlm = (const float*)d_in[18];
  const float* blm = (const float*)d_in[19];
  float* out = (float*)d_out;

  // workspace layout (fp32): x, h, q, k, v, o (M_*E_ each) + mid (M_*4E_)
  float* x   = (float*)d_ws;
  float* hb  = x  + (size_t)M_ * E_;
  float* qb  = hb + (size_t)M_ * E_;
  float* kb  = qb + (size_t)M_ * E_;
  float* vb  = kb + (size_t)M_ * E_;
  float* ob  = vb + (size_t)M_ * E_;
  float* mid = ob + (size_t)M_ * E_;

  dim3 blk(256);
  // blockIdx.x = M tile (fast) so concurrent blocks share weight stripes in L2
  dim3 gEE(M_ / 128, E_ / 128);            // N = 1024
  dim3 gE4(M_ / 128, 4 * E_ / 128);        // N = 4096
  dim3 gLM(M_ / 128, (V_ + 127) / 128);    // N = 50257
  dim3 gAT(T_ / 64, B_ * H_);

  embed_k<<<M_, blk, 0, stream>>>(idx, tok, pos, x);
  for (int l = 0; l < L_; ++l) {
    const size_t wo  = (size_t)l * E_ * E_;
    const size_t wo1 = (size_t)l * E_ * 4 * E_;
    layernorm_k<<<M_, blk, 0, stream>>>(x, g1 + l * E_, be1 + l * E_, hb);
    gemm_wmma<EPI_NONE><<<gEE, blk, 0, stream>>>(hb, Wq + wo, nullptr, nullptr, qb, M_, E_, E_);
    gemm_wmma<EPI_NONE><<<gEE, blk, 0, stream>>>(hb, Wk + wo, nullptr, nullptr, kb, M_, E_, E_);
    gemm_wmma<EPI_NONE><<<gEE, blk, 0, stream>>>(hb, Wv + wo, nullptr, nullptr, vb, M_, E_, E_);
    attn_wmma<<<gAT, blk, 0, stream>>>(qb, kb, vb, ob);
    gemm_wmma<EPI_BIAS_RES><<<gEE, blk, 0, stream>>>(ob, Wo + wo, bo + l * E_, x, x, M_, E_, E_);
    layernorm_k<<<M_, blk, 0, stream>>>(x, g2 + l * E_, be2 + l * E_, hb);
    gemm_wmma<EPI_BIAS_GELU><<<gE4, blk, 0, stream>>>(hb, W1 + wo1, b1 + (size_t)l * 4 * E_,
                                                      nullptr, mid, M_, 4 * E_, E_);
    gemm_wmma<EPI_BIAS_RES><<<gEE, blk, 0, stream>>>(mid, W2 + wo1, b2 + l * E_, x, x,
                                                     M_, E_, 4 * E_);
  }
  layernorm_k<<<M_, blk, 0, stream>>>(x, gf, bf, hb);
  gemm_wmma<EPI_BIAS><<<gLM, blk, 0, stream>>>(hb, Wlm, blm, nullptr, out, M_, V_, E_);
}
